// QRNN_48215302865709
// MI455X (gfx1250) — matrix operand order repaired and checked by reading, not compiled, not run on hardware
//
#include <hip/hip_runtime.h>
#include <hip/hip_bf16.h>

// ---------------------------------------------------------------------------
// Problem constants (from reference): B=16, L=2048, D0=512, O=512, K=3
// ---------------------------------------------------------------------------
#define BATCH 16
#define SEQL  2048
#define LPAD  2050          // L + 2 halo rows (conv padding)
#define D0    512
#define D1    1024          // 2*O (layer-1 input width)
#define GCH   1536          // 3*O gate channels
#define OCH   512

typedef __attribute__((ext_vector_type(16))) __bf16 v16bf;
typedef __attribute__((ext_vector_type(8)))  float  v8f;
typedef unsigned int u32x4 __attribute__((ext_vector_type(4)));

union Frag { v16bf v; u32x4 q[2]; };

__device__ __forceinline__ unsigned short f2bf(float f) {
  unsigned int u = __float_as_uint(f);
  u += 0x7FFFu + ((u >> 16) & 1u);          // round-to-nearest-even
  return (unsigned short)(u >> 16);
}

// ---------------------------------------------------------------------------
// Prep: fp32 x -> bf16 padded [B][LPAD][D0], zero halo rows
// ---------------------------------------------------------------------------
__global__ void __launch_bounds__(256)
pad_convert_x(const float* __restrict__ x, unsigned short* __restrict__ xp) {
  const long n = (long)BATCH * LPAD * D0;
  for (long i = (long)blockIdx.x * blockDim.x + threadIdx.x; i < n;
       i += (long)gridDim.x * blockDim.x) {
    long d = i % D0;
    long r = i / D0;
    long p = r % LPAD;
    long b = r / LPAD;
    unsigned short v = 0;
    if (p >= 1 && p <= SEQL)
      v = f2bf(x[(b * SEQL + (p - 1)) * D0 + d]);
    xp[i] = v;
  }
}

// zero the halo rows of the layer-1 padded input [B][LPAD][D1]
__global__ void __launch_bounds__(256)
zero_pad_x1(unsigned short* __restrict__ xp1) {
  int i = blockIdx.x * blockDim.x + threadIdx.x;
  if (i < BATCH * 2 * D1) {
    int d = i % D1;
    int r = i / D1;
    int b = r >> 1;
    int p = (r & 1) ? (LPAD - 1) : 0;
    xp1[((long)b * LPAD + p) * D1 + d] = 0;
  }
}

// weights [3O][Din][3] fp32  ->  bf16 [tap][n][kd]  (K contiguous per column)
__global__ void __launch_bounds__(256)
prep_w(const float* __restrict__ w, unsigned short* __restrict__ wp, int Din) {
  const long n = 3L * GCH * Din;
  for (long i = (long)blockIdx.x * blockDim.x + threadIdx.x; i < n;
       i += (long)gridDim.x * blockDim.x) {
    long kd   = i % Din;
    long rest = i / Din;
    long oc   = rest % GCH;
    long tap  = rest / GCH;
    wp[i] = f2bf(w[(oc * Din + kd) * 3 + tap]);
  }
}

// ---------------------------------------------------------------------------
// Conv-as-GEMM with WMMA bf16 (DIN compile-time; inner 4-step K-loop fully
// unrolled so all global_load_b128 carry immediate offsets):
//   gates[b*L+t][n] = bias[n] + sum_{tap,kd} xp[b][t+tap][kd] * wp[tap][n][kd]
// Block tile 256x128, 8 waves of 64x64 (4x4 v8f accumulators):
// per 32-deep K-step each wave issues 16 b128 loads for 16 WMMAs.
// Fragments per CDNA5 ISA bf16 layouts:
//   A 16x32: lane h=l>>4,m=l&15 holds K = h*8+[0..7] and 16+h*8+[0..7]
//   B 32x16: lane n=l&15       holds K = h*8+[0..7] and 16+h*8+[0..7]   (wp is [n][kd])
//   C 16x16: elem e -> M = h*8+e, N = l&15
// ---------------------------------------------------------------------------
template <int DIN>
__global__ void __launch_bounds__(256)
gemm_conv_wmma(const unsigned short* __restrict__ xp,   // [B][LPAD][DIN] bf16
               const unsigned short* __restrict__ wp,   // [3][GCH][DIN]  bf16
               const float*          __restrict__ bias, // [GCH]
               float*                __restrict__ gates)// [B*L][GCH]
{
  const int ntile = blockIdx.x;            // 12 tiles of 128 cols
  const int mtile = blockIdx.y;            // 128 tiles of 256 rows
  const int wave  = threadIdx.x >> 5;
  const int lane  = threadIdx.x & 31;
  const int wm    = wave & 3;              // 4 waves over M (64 rows each)
  const int wn    = wave >> 2;             // 2 waves over N (64 cols each)
  const int h     = lane >> 4;
  const int ln    = lane & 15;

  const int  R0    = mtile * 256;          // global output row base
  const int  batch = R0 / SEQL;            // 256 | 2048: tile never crosses b
  const int  t0    = R0 % SEQL;
  const int  rowA0 = wm * 64;
  const int  col0  = ntile * 128 + wn * 64;

  // per-lane fragment base pointers (hoisted; loops below add constants only)
  const unsigned short* ap =
      xp + ((long)batch * LPAD + t0 + rowA0 + ln) * DIN + h * 8;
  const unsigned short* bp =
      wp + (long)(col0 + ln) * DIN + h * 8;

  v8f acc[4][4];
#pragma unroll
  for (int i = 0; i < 4; ++i)
#pragma unroll
    for (int j = 0; j < 4; ++j)
      acc[i][j] = (v8f){0.f, 0.f, 0.f, 0.f, 0.f, 0.f, 0.f, 0.f};

  for (int tap = 0; tap < 3; ++tap) {
    for (int kb = 0; kb < DIN; kb += 128) {     // strength-reduced ptr bump
#pragma unroll
      for (int ku = 0; ku < 128; ku += 32) {    // immediate offsets
        const int k0 = kb + ku;
        Frag af[4];
#pragma unroll
        for (int i = 0; i < 4; ++i) {
          af[i].q[0] = *(const u32x4*)(ap + i * 16 * DIN + k0);       // K=h*8+[0..7]
          af[i].q[1] = *(const u32x4*)(ap + i * 16 * DIN + k0 + 16);  // K=16+h*8+[0..7]
        }
        Frag bfr[4];
#pragma unroll
        for (int j = 0; j < 4; ++j) {
          bfr[j].q[0] = *(const u32x4*)(bp + j * 16 * DIN + k0);      // K=h*8+[0..7]
          bfr[j].q[1] = *(const u32x4*)(bp + j * 16 * DIN + k0 + 16); // K=16+h*8+[0..7]
        }
#pragma unroll
        for (int i = 0; i < 4; ++i)
#pragma unroll
          for (int j = 0; j < 4; ++j)
            acc[i][j] = __builtin_amdgcn_wmma_f32_16x16x32_bf16(
                false, af[i].v, false, bfr[j].v, (short)0, acc[i][j],
                false, false);
      }
    }
    ap += DIN;                 // next tap = next padded row
    bp += (long)GCH * DIN;     // next tap's weight slab
  }

  // epilogue: add bias, scatter fp32 gates
#pragma unroll
  for (int j = 0; j < 4; ++j) {
    const int   n  = col0 + j * 16 + ln;
    const float bv = bias[n];
#pragma unroll
    for (int i = 0; i < 4; ++i) {
      const int mrow = R0 + rowA0 + i * 16 + h * 8;
      float* g = gates + (long)mrow * GCH + n;
#pragma unroll
      for (int e = 0; e < 8; ++e)
        g[(long)e * GCH] = acc[i][j][e] + bv;
    }
  }
}

// ---------------------------------------------------------------------------
// fo-pool: sequential scan per (batch, channel); one thread per lane.
//   c_t = sigma(f)*c_{t-1} + (1-sigma(f))*tanh(z);  h_t = c_t * sigma(o)
// Writes bf16 hidden into the padded next-layer input (layer 0) or fp32
// into out2 (layer 1); records last h/c into d_out hidden/cell sections.
// ---------------------------------------------------------------------------
__global__ void __launch_bounds__(128)
fo_pool_kernel(const float* __restrict__ gates,          // [B*L][GCH]
               unsigned short* __restrict__ xnext,       // [B][LPAD][D1] or null
               float* __restrict__ out2,                 // [B][L][D1] or null
               float* __restrict__ lastH, float* __restrict__ lastC,
               int reverse, int dirOff, int layerRow) {
  const int tid = blockIdx.x * blockDim.x + threadIdx.x;  // 0..8191
  const int b   = tid >> 9;
  const int ch  = tid & (OCH - 1);
  float c = 0.f;
  for (int s = 0; s < SEQL; ++s) {
    const int t = reverse ? (SEQL - 1 - s) : s;
    const float* g = gates + ((long)(b * SEQL + t)) * GCH + ch;
    const float f  = 1.f / (1.f + __expf(-g[0]));
    const float o  = 1.f / (1.f + __expf(-g[OCH]));
    const float z  = tanhf(g[2 * OCH]);
    c = f * c + (1.f - f) * z;
    const float hd = c * o;
    if (xnext)
      xnext[((long)b * LPAD + (t + 1)) * D1 + dirOff + ch] = f2bf(hd);
    if (out2)
      out2[((long)b * SEQL + t) * D1 + dirOff + ch] = hd;
    if (t == SEQL - 1) {  // fwd: final step; rev: first step (== rcells[:,-1])
      const long oi = (long)(layerRow * 16 + b) * D1 + dirOff + ch;
      lastH[oi] = hd;
      lastC[oi] = c;
    }
  }
}

// ---------------------------------------------------------------------------
// Host-side launcher
// ---------------------------------------------------------------------------
extern "C" void kernel_launch(void* const* d_in, const int* in_sizes, int n_in,
                              void* d_out, int out_size, void* d_ws,
                              size_t ws_size, hipStream_t stream) {
  const float* x   = (const float*)d_in[0];
  const float* w0f = (const float*)d_in[1];
  const float* b0f = (const float*)d_in[2];
  const float* w0r = (const float*)d_in[3];
  const float* b0r = (const float*)d_in[4];
  const float* w1f = (const float*)d_in[5];
  const float* b1f = (const float*)d_in[6];
  const float* w1r = (const float*)d_in[7];
  const float* b1r = (const float*)d_in[8];

  // workspace layout (bytes, all naturally 256-aligned)
  char* ws = (char*)d_ws;
  const size_t SZ_XP0 = (size_t)BATCH * LPAD * D0 * 2;   //  33.6 MB
  const size_t SZ_XP1 = (size_t)BATCH * LPAD * D1 * 2;   //  67.2 MB
  const size_t SZ_W0  = 3UL * GCH * D0 * 2;              //   4.7 MB
  const size_t SZ_W1  = 3UL * GCH * D1 * 2;              //   9.4 MB
  unsigned short* xp0  = (unsigned short*)(ws);
  unsigned short* xp1  = (unsigned short*)(ws + SZ_XP0);
  unsigned short* w0ft = (unsigned short*)(ws + SZ_XP0 + SZ_XP1);
  unsigned short* w0rt = (unsigned short*)(ws + SZ_XP0 + SZ_XP1 + SZ_W0);
  unsigned short* w1ft = (unsigned short*)(ws + SZ_XP0 + SZ_XP1 + 2 * SZ_W0);
  unsigned short* w1rt = (unsigned short*)(ws + SZ_XP0 + SZ_XP1 + 2 * SZ_W0 + SZ_W1);
  float* gates = (float*)(ws + SZ_XP0 + SZ_XP1 + 2 * SZ_W0 + 2 * SZ_W1); // 192 MB

  float* out2  = (float*)d_out;                          // [B][L][2O]
  float* lastH = out2 + (long)BATCH * SEQL * D1;         // [4][B][O] flat
  float* lastC = lastH + 4L * BATCH * OCH;

  // prep (independent)
  pad_convert_x<<<2048, 256, 0, stream>>>(x, xp0);
  zero_pad_x1<<<(BATCH * 2 * D1 + 255) / 256, 256, 0, stream>>>(xp1);
  prep_w<<<1024, 256, 0, stream>>>(w0f, w0ft, D0);
  prep_w<<<1024, 256, 0, stream>>>(w0r, w0rt, D0);
  prep_w<<<2048, 256, 0, stream>>>(w1f, w1ft, D1);
  prep_w<<<2048, 256, 0, stream>>>(w1r, w1rt, D1);

  const dim3 gg(GCH / 128, (BATCH * SEQL) / 256);        // 12 x 128 blocks

  // layer 0, forward
  gemm_conv_wmma<D0><<<gg, 256, 0, stream>>>(xp0, w0ft, b0f, gates);
  fo_pool_kernel<<<64, 128, 0, stream>>>(gates, xp1, nullptr, lastH, lastC,
                                         0, 0, 0);
  // layer 0, reverse
  gemm_conv_wmma<D0><<<gg, 256, 0, stream>>>(xp0, w0rt, b0r, gates);
  fo_pool_kernel<<<64, 128, 0, stream>>>(gates, xp1, nullptr, lastH, lastC,
                                         1, OCH, 0);
  // layer 1, forward
  gemm_conv_wmma<D1><<<gg, 256, 0, stream>>>(xp1, w1ft, b1f, gates);
  fo_pool_kernel<<<64, 128, 0, stream>>>(gates, nullptr, out2, lastH, lastC,
                                         0, 0, 1);
  // layer 1, reverse
  gemm_conv_wmma<D1><<<gg, 256, 0, stream>>>(xp1, w1rt, b1r, gates);
  fo_pool_kernel<<<64, 128, 0, stream>>>(gates, nullptr, out2, lastH, lastC,
                                         1, OCH, 1);
}
